// RNN_12558484373616
// MI455X (gfx1250) — matrix operand order repaired and checked by reading, not compiled, not run on hardware
//
#include <hip/hip_runtime.h>
#include <math.h>

// ---------------------------------------------------------------------------
// tanh-RNN for MI455X (gfx1250, wave32, WMMA)
//   Phase 1: xp = x @ W_ih^T + b_ih + b_hh    (f16 WMMA GEMM, f32 accum)
//   Phase 2: h_t = tanh(xp_t + h_{t-1} @ W_hh^T)  (persistent, 16 WGs,
//            W_hh slice resident in LDS via TDM, f32 WMMA 16x16x4 GEMV)
// ---------------------------------------------------------------------------

#define SEQ     16384
#define HID     1024
#define IN_DIM  1024
#define NWG     16            // persistent workgroups in recurrence
#define RPW     64            // rows of h per workgroup (HID / NWG)

typedef __attribute__((ext_vector_type(16))) _Float16 v16h;
typedef __attribute__((ext_vector_type(8)))  float    v8f;
typedef __attribute__((ext_vector_type(2)))  float    v2f;
typedef __attribute__((ext_vector_type(4)))  unsigned v4u;
typedef __attribute__((ext_vector_type(8)))  int      v8i;
typedef __attribute__((ext_vector_type(4)))  int      v4i;

// ---------------------------------------------------------------------------
// init: zero h0 (both ping-pong h buffers) and the grid-barrier state
// ---------------------------------------------------------------------------
__global__ void k_init(float* hbuf, unsigned* bar) {
  int i = blockIdx.x * blockDim.x + threadIdx.x;
  if (i < 2 * HID) hbuf[i] = 0.0f;
  if (i < 4)       bar[i]  = 0u;
}

// ---------------------------------------------------------------------------
// Pack W_ih (fp32 [H, K] row-major) into f16 WMMA B-fragment order.
// B 32x16 f16 order used here (contiguous k, not the ISA table order): lane L: N = L%16, halves h: K = (L>>4)*16 + h.
// Packed index p = ((nt*32 + kt)*32 + L)*16 + h.
// ---------------------------------------------------------------------------
__global__ void k_pack_wih(const float* __restrict__ wih, _Float16* __restrict__ out) {
  unsigned p  = blockIdx.x * 256u + threadIdx.x;     // 0 .. 2^20-1
  unsigned h  = p & 15u;
  unsigned L  = (p >> 4) & 31u;
  unsigned kt = (p >> 9) & 31u;
  unsigned nt = p >> 14;
  unsigned n  = nt * 16u + (L & 15u);
  unsigned k  = kt * 32u + (L >> 4) * 16u + h;
  out[p] = (_Float16)wih[n * IN_DIM + k];
}

// ---------------------------------------------------------------------------
// Pack W_hh (fp32 [H, K]) into fp32 WMMA A-fragment (16x4) order.
// A 16x4 f32 layout: lane L: M = L%16; VGPR0/1 = K = (L>>4)*2 + {0,1}.
// Packed pair-element index q = (((mtg*256 + kt)*32 + L)*2 + e);  mtg = wg*4+mt.
// ---------------------------------------------------------------------------
__global__ void k_pack_whh(const float* __restrict__ whh, float* __restrict__ out) {
  unsigned q   = blockIdx.x * 256u + threadIdx.x;    // 0 .. 2^20-1
  unsigned e   = q & 1u;
  unsigned L   = (q >> 1) & 31u;
  unsigned kt  = (q >> 6) & 255u;
  unsigned mtg = q >> 14;                            // 0..63
  unsigned row = mtg * 16u + (L & 15u);
  unsigned k   = kt * 4u + (L >> 4) * 2u + e;
  out[q] = whh[row * HID + k];
}

// ---------------------------------------------------------------------------
// Phase 1 GEMM: each wave owns one 16x16 tile of xp; K-loop of 32 f16 WMMAs.
// A (x) converted fp32->f16 inline; A 16x32 layout: lane L: M=L%16,
// halves 0..15: K = (L>>4)*16 + h (matches packed B).
// Output written to d_out (xp lives in-place there until phase 2 consumes it).
// ---------------------------------------------------------------------------
__global__ void __launch_bounds__(256) k_gemm_xp(
    const float* __restrict__ x, const _Float16* __restrict__ wihp,
    const float* __restrict__ b_ih, const float* __restrict__ b_hh,
    float* __restrict__ out) {
  const int wid  = blockIdx.x * 8 + (threadIdx.x >> 5);
  const int mt   = wid >> 6;                // 0..1023 (SEQ/16)
  const int nt   = wid & 63;                // 0..63  (HID/16)
  const int lane = threadIdx.x & 31;
  const int agrp = lane >> 4;

  const float* xrow = x + (size_t)(mt * 16 + (lane & 15)) * IN_DIM;
  v8f acc = {};
  for (int kt = 0; kt < 32; ++kt) {
    const float4* ap = (const float4*)(xrow + kt * 32 + agrp * 16);
    float4 a0 = ap[0], a1 = ap[1], a2 = ap[2], a3 = ap[3];
    v16h A;
    A[0]=(_Float16)a0.x;  A[1]=(_Float16)a0.y;  A[2]=(_Float16)a0.z;  A[3]=(_Float16)a0.w;
    A[4]=(_Float16)a1.x;  A[5]=(_Float16)a1.y;  A[6]=(_Float16)a1.z;  A[7]=(_Float16)a1.w;
    A[8]=(_Float16)a2.x;  A[9]=(_Float16)a2.y;  A[10]=(_Float16)a2.z; A[11]=(_Float16)a2.w;
    A[12]=(_Float16)a3.x; A[13]=(_Float16)a3.y; A[14]=(_Float16)a3.z; A[15]=(_Float16)a3.w;
    v16h B = *(const v16h*)(wihp + ((size_t)(nt * 32 + kt) * 32 + lane) * 16);
    acc = __builtin_amdgcn_wmma_f32_16x16x32_f16(false, A, false, B,
                                                 (short)0, acc, false, false);
  }
  // D layout: VGPR r, lane L -> M = r + (L>=16)*8, N = L%16
  const int   col  = nt * 16 + (lane & 15);
  const float bias = b_ih[col] + b_hh[col];
  const int   mb   = mt * 16 + agrp * 8;
#pragma unroll
  for (int r = 0; r < 8; ++r)
    out[(size_t)(mb + r) * HID + col] = acc[r] + bias;
}

// ---------------------------------------------------------------------------
// Phase 2: persistent recurrence. 16 blocks x 128 threads (4 waves).
// Wave w computes output rows [wg*64 + w*16, +16) each timestep from its
// LDS-resident W_hh slice (staged once via the Tensor Data Mover).
// ---------------------------------------------------------------------------
__global__ void __launch_bounds__(128) k_rnn(
    float* __restrict__ out,            // in: xp, out: h (in place per row)
    const float* __restrict__ whhp,     // packed W_hh fragments
    float* __restrict__ hbuf,           // 2 x HID f32 ping-pong
    unsigned* __restrict__ bar) {       // [0]=count, [1]=generation
  extern __shared__ float smem[];
  float* lw = smem;                     // 64*1024 f32 = 256 KB weight slice
  float* hs = smem + RPW * HID;         // 1024 f32 staged h_t

  const int wg   = blockIdx.x;
  const int tid  = threadIdx.x;
  const int wave = tid >> 5;
  const int lane = tid & 31;
  const int agrp = lane >> 4;
  const int mrow = lane & 15;

  // ---- stage this WG's 256 KB W_hh slice into LDS with the TDM ----
  if (wave == 0) {
    unsigned ldsoff = __builtin_amdgcn_groupstaticsize();  // dynamic-LDS base
    unsigned long long ga =
        (unsigned long long)(const void*)(whhp + (size_t)wg * (RPW * HID));
    v4u g0 = {0u, 0u, 0u, 0u};
    g0[0] = 1u;                                            // count = 1
    g0[1] = ldsoff;                                        // lds_addr
    g0[2] = (unsigned)ga;                                  // global_addr lo
    g0[3] = (unsigned)((ga >> 32) & 0x1FFFFFFull) | (2u << 30);  // hi + type=2
    v8i g1 = {0, 0, 0, 0, 0, 0, 0, 0};
    g1[0] = (int)(2u << 16);        // data_size = 4 bytes
    g1[1] = (int)(32768u << 16);    // tensor_dim0 = 32768 (bits 79:48 lo)
    g1[2] = (int)(2u << 16);        // tensor_dim1 = 2     (bits 111:80 lo)
    g1[3] = (int)(32768u << 16);    // tile_dim0 = 32768   (bits 127:112)
    g1[4] = 2;                      // tile_dim1 = 2
    g1[5] = 32768;                  // tensor_dim0_stride = 32768
    v4i gz = {0, 0, 0, 0};
    v8i gz8 = {0, 0, 0, 0, 0, 0, 0, 0};
#if __has_builtin(__builtin_amdgcn_tensor_load_to_lds)
    __builtin_amdgcn_tensor_load_to_lds(g0, g1, gz, gz, gz8, 0);
    __builtin_amdgcn_s_wait_tensorcnt(0);
#else
    (void)gz8;
    for (int i = lane; i < RPW * HID; i += 32)
      lw[i] = whhp[(size_t)wg * (RPW * HID) + i];
#endif
  }
  __syncthreads();

  unsigned* cnt = bar;
  unsigned* gen = bar + 1;
  const v2f* aw = ((const v2f*)lw) + (size_t)wave * (256 * 32) + lane;

#pragma unroll 1
  for (int t = 0; t < SEQ; ++t) {
    const float* hsrc = hbuf + (t & 1) * HID;
    float*       hdst = hbuf + ((t + 1) & 1) * HID;

    // stage h_t (4 KB) into LDS, shared by all 4 waves
    for (int i = tid; i < HID / 4; i += 128)
      ((float4*)hs)[i] = ((const float4*)hsrc)[i];
    __syncthreads();

    // GEMV slice: D(16x16, all columns identical) = W_hh(16xK) x h(Kx16 bcast)
    v8f c0 = {}, c1 = {}, c2 = {}, c3 = {};
#pragma unroll 4
    for (int kt = 0; kt < 256; kt += 4) {
      v2f b0 = *(const v2f*)(hs + (kt + 0) * 4 + agrp * 2);
      v2f b1 = *(const v2f*)(hs + (kt + 1) * 4 + agrp * 2);
      v2f b2 = *(const v2f*)(hs + (kt + 2) * 4 + agrp * 2);
      v2f b3 = *(const v2f*)(hs + (kt + 3) * 4 + agrp * 2);
      c0 = __builtin_amdgcn_wmma_f32_16x16x4_f32(false, aw[(kt + 0) * 32], false, b0,
                                                 (short)0, c0, false, false);
      c1 = __builtin_amdgcn_wmma_f32_16x16x4_f32(false, aw[(kt + 1) * 32], false, b1,
                                                 (short)0, c1, false, false);
      c2 = __builtin_amdgcn_wmma_f32_16x16x4_f32(false, aw[(kt + 2) * 32], false, b2,
                                                 (short)0, c2, false, false);
      c3 = __builtin_amdgcn_wmma_f32_16x16x4_f32(false, aw[(kt + 3) * 32], false, b3,
                                                 (short)0, c3, false, false);
    }
    v8f c = (c0 + c1) + (c2 + c3);

    // Lanes with N==0 (lanes 0 and 16) own M = agrp*8 + r in VGPR r.
    if (mrow == 0) {
      const int ob   = wg * RPW + wave * 16 + agrp * 8;
      float*    orow = out + (size_t)t * HID + ob;
      if (t + 1 < SEQ)
        __builtin_prefetch(out + (size_t)(t + 1) * HID + ob, 0, 3);
#pragma unroll
      for (int r = 0; r < 8; ++r) {
        float hv = tanhf(c[r] + orow[r]);   // read xp, then overwrite with h
        orow[r]      = hv;
        hdst[ob + r] = hv;
      }
    }

    // ---- device-wide barrier between timesteps ----
    __syncthreads();
    if (tid == 0) {
      __builtin_amdgcn_s_cluster_barrier();  // NOP when not a cluster dispatch
      __threadfence();
      unsigned prev = atomicAdd(cnt, 1u);
      if (prev == NWG - 1) {
        atomicExch(cnt, 0u);
        __threadfence();
        atomicExch(gen, (unsigned)(t + 1));
      } else {
        while (__hip_atomic_load(gen, __ATOMIC_RELAXED, __HIP_MEMORY_SCOPE_AGENT) <
               (unsigned)(t + 1)) {
          __builtin_amdgcn_s_sleep(8);
        }
      }
      __threadfence();
    }
    __syncthreads();
  }
}

// ---------------------------------------------------------------------------
extern "C" void kernel_launch(void* const* d_in, const int* in_sizes, int n_in,
                              void* d_out, int out_size, void* d_ws, size_t ws_size,
                              hipStream_t stream) {
  (void)in_sizes; (void)n_in; (void)out_size; (void)ws_size;
  const float* x   = (const float*)d_in[0];
  const float* Wih = (const float*)d_in[1];
  const float* Whh = (const float*)d_in[2];
  const float* bih = (const float*)d_in[3];
  const float* bhh = (const float*)d_in[4];
  float* out = (float*)d_out;

  char* ws = (char*)d_ws;
  _Float16* wih_pk = (_Float16*)ws;                        // 2 MB
  float*    whh_pk = (float*)(ws + (2u << 20));            // 4 MB
  float*    hbuf   = (float*)(ws + (6u << 20));            // 8 KB
  unsigned* bar    = (unsigned*)(ws + (6u << 20) + 8192);  // barrier state

  k_init<<<8, 256, 0, stream>>>(hbuf, bar);
  k_pack_wih<<<4096, 256, 0, stream>>>(Wih, wih_pk);
  k_pack_whh<<<4096, 256, 0, stream>>>(Whh, whh_pk);
  k_gemm_xp<<<8192, 256, 0, stream>>>(x, wih_pk, bih, bhh, out);

  size_t lds_bytes = (size_t)(RPW * HID + HID) * sizeof(float);  // 260 KB
  k_rnn<<<NWG, 128, lds_bytes, stream>>>(out, whh_pk, hbuf, bar);
}
